// DSQGAttentionTCR_25451976196242
// MI455X (gfx1250) — compile-verified, matrix operand order checked by reading, not run on hardware
//
#include <hip/hip_runtime.h>
#include <hip/hip_bf16.h>

typedef _Float16 h16;
typedef _Float16 v8h  __attribute__((ext_vector_type(8)));
typedef _Float16 v16h __attribute__((ext_vector_type(16)));
typedef float    v8f  __attribute__((ext_vector_type(8)));

#define BB  4
#define NN  2048
#define DD  512
#define HH  8
#define HDIM 64
#define TOK (BB * NN)          // 8192 tokens
#define D3  (3 * DD)           // 1536

// ---------------------------------------------------------------------------
// Fragment loader. For WMMA f16 A (16x32, MxK) the per-lane layout is:
//   lane = m + 16*g ; halves 0..7 <- K = g*8..g*8+7 ; halves 8..15 <- K = 16+g*8..
// B (32x16, KxN) mirrors this with lane = n. For a row-major [16][K] source
// (A itself, or B^T), each lane does two contiguous 16B loads.
// ---------------------------------------------------------------------------
__device__ __forceinline__ v16h ldfrag(const h16* base, int stride, int row,
                                       int kbase, int g) {
  const h16* p = base + (size_t)row * stride + kbase + g * 8;
  v8h lo = *(const v8h*)(p);
  v8h hi = *(const v8h*)(p + 16);
  v16h r;
#pragma unroll
  for (int i = 0; i < 8; ++i) { r[i] = lo[i]; r[i + 8] = hi[i]; }
  return r;
}

__device__ __forceinline__ v8f wmma32(v16h a, v16h b, v8f c) {
  return __builtin_amdgcn_wmma_f32_16x16x32_f16(false, a, false, b, (short)0, c,
                                                false, false);
}

// 16-lane (half-wave) reductions; xor masks 1,2,4,8 stay inside each half.
__device__ __forceinline__ float redmax16(float x) {
#pragma unroll
  for (int off = 1; off < 16; off <<= 1) x = fmaxf(x, __shfl_xor(x, off, 32));
  return x;
}
__device__ __forceinline__ float redsum16(float x) {
#pragma unroll
  for (int off = 1; off < 16; off <<= 1) x += __shfl_xor(x, off, 32);
  return x;
}

// ---------------------------------------------------------------------------
// Prep kernels
// ---------------------------------------------------------------------------
__global__ void cvt_f16(const float* __restrict__ in, h16* __restrict__ out, int n) {
  int i = blockIdx.x * blockDim.x + threadIdx.x;
  if (i < n) out[i] = (h16)in[i];
}

// in[K][Nout] f32  ->  out[Nout][K] f16
__global__ void transpose_w(const float* __restrict__ in, h16* __restrict__ out,
                            int K, int Nout) {
  int i = blockIdx.x * blockDim.x + threadIdx.x;
  if (i < K * Nout) {
    int k = i / Nout, o = i % Nout;
    out[(size_t)o * K + k] = (h16)in[i];
  }
}

// tab[h][dist] = in-band ? pos_bias[bucket][h] : -100   (T5 buckets, d = i-j >= 0)
__global__ void build_bias_tab(const float* __restrict__ pos_bias,
                               float* __restrict__ tab) {
  int i = blockIdx.x * blockDim.x + threadIdx.x;
  if (i >= HH * NN) return;
  int h = i >> 11, d = i & (NN - 1);
  int bucket;
  if (d < 28) {
    bucket = d;
  } else {
    float lb = logf((float)d * (1.0f / 28.0f)) * (16.0f / logf(2048.0f / 28.0f));
    bucket = 28 + (int)lb;
    if (bucket > 43) bucket = 43;
  }
  const int bs[8] = {38, 34, 28, 20, 13, 7, 3, 0};
  const int be[8] = {44, 40, 35, 28, 21, 14, 9, 6};
  tab[i] = (bucket >= bs[h] && bucket < be[h]) ? pos_bias[bucket * HH + h] : -100.0f;
}

// ---------------------------------------------------------------------------
// QKV projection: [8192,512] x [512,1536] via WMMA f16->f32.
// One wave = 16x64 output tile (4 C tiles). 16 K-steps, fully unrolled
// double buffer; sched_barrier(0) pins each load group ABOVE the previous
// group's WMMAs so one full load group is always in flight (wait != 0).
// ---------------------------------------------------------------------------
__global__ __launch_bounds__(128) void qkv_gemm(
    const h16* __restrict__ x16, const h16* __restrict__ wT,
    const float* __restrict__ bias, const float* __restrict__ k_delta,
    const float* __restrict__ v_delta, h16* __restrict__ q16,
    h16* __restrict__ k16, h16* __restrict__ vt16) {
  int wid  = (blockIdx.x * blockDim.x + threadIdx.x) >> 5;
  int lane = threadIdx.x & 31;
  const int COLT = D3 / 64;                       // 24
  if (wid >= (TOK / 16) * COLT) return;
  int n0 = (wid / COLT) * 16, o0 = (wid % COLT) * 64;
  int g = lane >> 4, lr = lane & 15;

  const h16* xr = x16 + (size_t)n0 * DD;
  const h16* w0 = wT + (size_t)o0 * DD;
  v8f z = {};
  v8f acc[4] = {z, z, z, z};

  v16h a[2];
  v16h b[2][4];
  a[0] = ldfrag(xr, DD, lr, 0, g);
#pragma unroll
  for (int t = 0; t < 4; ++t) b[0][t] = ldfrag(w0 + (size_t)t * 16 * DD, DD, lr, 0, g);

#pragma unroll
  for (int s = 0; s < DD / 32; ++s) {
    int cur = s & 1, nxt = cur ^ 1;
    int cn = (s + 1) * 32;
    if (cn < DD) {
      a[nxt] = ldfrag(xr, DD, lr, cn, g);
#pragma unroll
      for (int t = 0; t < 4; ++t)
        b[nxt][t] = ldfrag(w0 + (size_t)t * 16 * DD, DD, lr, cn, g);
    }
    __builtin_amdgcn_sched_barrier(0);
#pragma unroll
    for (int t = 0; t < 4; ++t) acc[t] = wmma32(a[cur], b[cur][t], acc[t]);
  }

#pragma unroll
  for (int t = 0; t < 4; ++t) {
    int o = o0 + t * 16 + lr;
    int sec = o >> 9, f = o & 511, h = f >> 6, d = f & 63;
#pragma unroll
    for (int r = 0; r < 8; ++r) {
      int m = r + 8 * g;
      int tok = n0 + m;
      int b_ = tok >> 11, n = tok & (NN - 1);
      float v = acc[t][r] + bias[o];
      size_t hidx = (((size_t)b_ * HH + h) * NN + n) * HDIM + d;
      if (sec == 0) {
        q16[hidx] = (h16)v;
      } else if (sec == 1) {
        k16[hidx] = (h16)(v + k_delta[hidx]);
      } else {
        vt16[(((size_t)b_ * HH + h) * HDIM + d) * NN + n] = (h16)(v + v_delta[hidx]);
      }
    }
  }
}

// ---------------------------------------------------------------------------
// Gate projection: sigmoid(x @ gate_w + b) -> f16 [TOK, D].
// ---------------------------------------------------------------------------
__global__ __launch_bounds__(128) void gate_gemm(
    const h16* __restrict__ x16, const h16* __restrict__ wT,
    const float* __restrict__ bias, h16* __restrict__ gate16) {
  int wid  = (blockIdx.x * blockDim.x + threadIdx.x) >> 5;
  int lane = threadIdx.x & 31;
  const int COLT = DD / 64;                       // 8
  if (wid >= (TOK / 16) * COLT) return;
  int n0 = (wid / COLT) * 16, o0 = (wid % COLT) * 64;
  int g = lane >> 4, lr = lane & 15;

  const h16* xr = x16 + (size_t)n0 * DD;
  const h16* w0 = wT + (size_t)o0 * DD;
  v8f z = {};
  v8f acc[4] = {z, z, z, z};

  v16h a[2];
  v16h b[2][4];
  a[0] = ldfrag(xr, DD, lr, 0, g);
#pragma unroll
  for (int t = 0; t < 4; ++t) b[0][t] = ldfrag(w0 + (size_t)t * 16 * DD, DD, lr, 0, g);

#pragma unroll
  for (int s = 0; s < DD / 32; ++s) {
    int cur = s & 1, nxt = cur ^ 1;
    int cn = (s + 1) * 32;
    if (cn < DD) {
      a[nxt] = ldfrag(xr, DD, lr, cn, g);
#pragma unroll
      for (int t = 0; t < 4; ++t)
        b[nxt][t] = ldfrag(w0 + (size_t)t * 16 * DD, DD, lr, cn, g);
    }
    __builtin_amdgcn_sched_barrier(0);
#pragma unroll
    for (int t = 0; t < 4; ++t) acc[t] = wmma32(a[cur], b[cur][t], acc[t]);
  }

#pragma unroll
  for (int t = 0; t < 4; ++t) {
    int o = o0 + t * 16 + lr;
#pragma unroll
    for (int r = 0; r < 8; ++r) {
      int tok = n0 + r + 8 * g;
      float v = acc[t][r] + bias[o];
      gate16[(size_t)tok * DD + o] = (h16)(1.0f / (1.0f + __expf(-v)));
    }
  }
}

// ---------------------------------------------------------------------------
// Flash attention. One wave per (b, h, 16-row q tile); key blocks of 32.
// All K and V fragments of a block are issued before any WMMA (pinned by a
// sched_barrier so V loads stay hoisted): V-load latency hides behind the
// S WMMAs + softmax. P is transposed C->A layout through padded LDS with an
// explicit s_wait_dscnt. P x V is one K=32 WMMA per 16-wide d chunk.
// ---------------------------------------------------------------------------
#define PST 40   // LDS row stride in halves (80B): conflict-free, 16B aligned

__global__ __launch_bounds__(128) void attn_kernel(
    const h16* __restrict__ q16, const h16* __restrict__ k16,
    const h16* __restrict__ vt16, const float* __restrict__ tab,
    const float* __restrict__ if_gain, h16* __restrict__ y16) {
  __shared__ h16 lds[4 * 16 * PST];
  int lane  = threadIdx.x & 31;
  int wslot = threadIdx.x >> 5;
  int wid   = (blockIdx.x * blockDim.x + threadIdx.x) >> 5;   // 0..4095
  if (wid >= BB * HH * (NN / 16)) return;
  int b_ = wid >> 10;
  int h  = (wid >> 7) & (HH - 1);
  int i0 = (wid & 127) * 16;
  int g = lane >> 4, lr = lane & 15;

  const h16* qbh = q16 + ((size_t)b_ * HH + h) * NN * HDIM;
  const h16* kbh = k16 + ((size_t)b_ * HH + h) * NN * HDIM;
  const h16* vbh = vt16 + ((size_t)b_ * HH + h) * HDIM * NN;
  const float* tb = tab + h * NN;
  h16* P = lds + wslot * 16 * PST;

  v16h qa0 = ldfrag(qbh, HDIM, i0 + lr, 0, g);
  v16h qa1 = ldfrag(qbh, HDIM, i0 + lr, 32, g);

  v8f z = {};
  v8f oacc[4] = {z, z, z, z};
  float mrow[8], lrow[8];
#pragma unroll
  for (int r = 0; r < 8; ++r) { mrow[r] = -3.0e38f; lrow[r] = 0.0f; }

  for (int j0 = 0; j0 <= i0; j0 += 32) {
    if (j0 + 32 <= i0)  // prefetch next K block (global_prefetch_b8)
      __builtin_prefetch(kbh + (size_t)(j0 + 32 + lane) * HDIM, 0, 1);

    // Issue ALL loads for this block up front (4 K frags + 4 V frags).
    v16h kb0 = ldfrag(kbh, HDIM, j0 + lr, 0, g);
    v16h kb1 = ldfrag(kbh, HDIM, j0 + lr, 32, g);
    v16h kb2 = ldfrag(kbh, HDIM, j0 + 16 + lr, 0, g);
    v16h kb3 = ldfrag(kbh, HDIM, j0 + 16 + lr, 32, g);
    v16h vf[4];
#pragma unroll
    for (int t = 0; t < 4; ++t)
      vf[t] = ldfrag(vbh, NN, t * 16 + lr, j0, g);
    __builtin_amdgcn_sched_barrier(0);   // keep V loads issued up here

    v8f sa = z, sb = z;
    sa = wmma32(qa0, kb0, sa);
    sa = wmma32(qa1, kb1, sa);
    sb = wmma32(qa0, kb2, sb);
    sb = wmma32(qa1, kb3, sb);

    int ja = j0 + lr, jb = j0 + 16 + lr;
#pragma unroll
    for (int r = 0; r < 8; ++r) {
      int m = r + 8 * g;
      int i = i0 + m;
      int da = i - ja, db = i - jb;
      float va = (da >= 0) ? sa[r] * 0.125f + tb[da] : -1.0e9f;
      float vb = (db >= 0) ? sb[r] * 0.125f + tb[db] : -1.0e9f;
      float mnew  = fmaxf(mrow[r], redmax16(fmaxf(va, vb)));
      float alpha = __expf(mrow[r] - mnew);
      float pa = __expf(va - mnew);
      float pb = __expf(vb - mnew);
      lrow[r] = lrow[r] * alpha + redsum16(pa + pb);
      mrow[r] = mnew;
#pragma unroll
      for (int t = 0; t < 4; ++t) oacc[t][r] *= alpha;
      P[m * PST + lr]      = (h16)pa;
      P[m * PST + 16 + lr] = (h16)pb;
    }
    __builtin_amdgcn_wave_barrier();
    asm volatile("s_wait_dscnt 0" ::: "memory");  // LDS stores visible to reload
    v16h pf = ldfrag(P, PST, lr, 0, g);
#pragma unroll
    for (int t = 0; t < 4; ++t) oacc[t] = wmma32(pf, vf[t], oacc[t]);
    __builtin_amdgcn_wave_barrier();              // keep next stores after loads
  }

  float gain = if_gain[h];
#pragma unroll
  for (int r = 0; r < 8; ++r) {
    int m = r + 8 * g;
    size_t tok = (size_t)b_ * NN + i0 + m;
    float s = gain / lrow[r];
#pragma unroll
    for (int t = 0; t < 4; ++t)
      y16[tok * DD + h * HDIM + t * 16 + lr] = (h16)(oacc[t][r] * s);
  }
}

// ---------------------------------------------------------------------------
// Final projection: out = (y * gate) @ out_w + out_b   (f32 output).
// Gate multiply fused into the A fragment in f16.
// ---------------------------------------------------------------------------
__global__ __launch_bounds__(128) void out_gemm(
    const h16* __restrict__ y16, const h16* __restrict__ gate16,
    const h16* __restrict__ wT, const float* __restrict__ bias,
    float* __restrict__ out) {
  int wid  = (blockIdx.x * blockDim.x + threadIdx.x) >> 5;
  int lane = threadIdx.x & 31;
  const int COLT = DD / 64;
  if (wid >= (TOK / 16) * COLT) return;
  int n0 = (wid / COLT) * 16, o0 = (wid % COLT) * 64;
  int g = lane >> 4, lr = lane & 15;

  const h16* yr = y16 + (size_t)n0 * DD;
  const h16* gr = gate16 + (size_t)n0 * DD;
  const h16* w0 = wT + (size_t)o0 * DD;
  v8f z = {};
  v8f acc[4] = {z, z, z, z};

  v16h ya[2], ga[2];
  v16h b[2][4];
  ya[0] = ldfrag(yr, DD, lr, 0, g);
  ga[0] = ldfrag(gr, DD, lr, 0, g);
#pragma unroll
  for (int t = 0; t < 4; ++t) b[0][t] = ldfrag(w0 + (size_t)t * 16 * DD, DD, lr, 0, g);

#pragma unroll
  for (int s = 0; s < DD / 32; ++s) {
    int cur = s & 1, nxt = cur ^ 1;
    int cn = (s + 1) * 32;
    if (cn < DD) {
      ya[nxt] = ldfrag(yr, DD, lr, cn, g);
      ga[nxt] = ldfrag(gr, DD, lr, cn, g);
#pragma unroll
      for (int t = 0; t < 4; ++t)
        b[nxt][t] = ldfrag(w0 + (size_t)t * 16 * DD, DD, lr, cn, g);
    }
    __builtin_amdgcn_sched_barrier(0);
    v16h afr;
#pragma unroll
    for (int i = 0; i < 16; ++i) afr[i] = (h16)(ya[cur][i] * ga[cur][i]);
#pragma unroll
    for (int t = 0; t < 4; ++t) acc[t] = wmma32(afr, b[cur][t], acc[t]);
  }

#pragma unroll
  for (int t = 0; t < 4; ++t) {
    int o = o0 + t * 16 + lr;
#pragma unroll
    for (int r = 0; r < 8; ++r) {
      int tok = n0 + r + 8 * g;
      out[(size_t)tok * DD + o] = acc[t][r] + bias[o];
    }
  }
}

// ---------------------------------------------------------------------------
// Launch
// ---------------------------------------------------------------------------
extern "C" void kernel_launch(void* const* d_in, const int* in_sizes, int n_in,
                              void* d_out, int out_size, void* d_ws, size_t ws_size,
                              hipStream_t stream) {
  const float* x        = (const float*)d_in[0];
  const float* k_delta  = (const float*)d_in[1];
  const float* v_delta  = (const float*)d_in[2];
  const float* qkv_w    = (const float*)d_in[3];
  const float* qkv_b    = (const float*)d_in[4];
  const float* gate_w   = (const float*)d_in[5];
  const float* gate_b   = (const float*)d_in[6];
  const float* out_w    = (const float*)d_in[7];
  const float* out_b    = (const float*)d_in[8];
  const float* pos_bias = (const float*)d_in[9];
  const float* if_gain  = (const float*)d_in[10];

  char* w = (char*)d_ws;
  auto alloc = [&](size_t bytes) -> char* {
    char* p = w;
    w += (bytes + 255) & ~(size_t)255;
    return p;
  };
  h16*   x16    = (h16*)alloc((size_t)TOK * DD * 2);          // 8 MB
  h16*   qkvwT  = (h16*)alloc((size_t)D3 * DD * 2);           // 1.5 MB
  h16*   gatewT = (h16*)alloc((size_t)DD * DD * 2);           // 0.5 MB
  h16*   outwT  = (h16*)alloc((size_t)DD * DD * 2);           // 0.5 MB
  h16*   q16    = (h16*)alloc((size_t)BB * HH * NN * HDIM * 2);
  h16*   k16    = (h16*)alloc((size_t)BB * HH * NN * HDIM * 2);
  h16*   vt16   = (h16*)alloc((size_t)BB * HH * NN * HDIM * 2);
  h16*   gate16 = (h16*)alloc((size_t)TOK * DD * 2);
  h16*   y16    = (h16*)alloc((size_t)TOK * DD * 2);
  float* tab    = (float*)alloc((size_t)HH * NN * 4);

  // 1) conversions / transposes / bias table
  cvt_f16<<<(TOK * DD + 255) / 256, 256, 0, stream>>>(x, x16, TOK * DD);
  transpose_w<<<(DD * D3 + 255) / 256, 256, 0, stream>>>(qkv_w, qkvwT, DD, D3);
  transpose_w<<<(DD * DD + 255) / 256, 256, 0, stream>>>(gate_w, gatewT, DD, DD);
  transpose_w<<<(DD * DD + 255) / 256, 256, 0, stream>>>(out_w, outwT, DD, DD);
  build_bias_tab<<<(HH * NN + 255) / 256, 256, 0, stream>>>(pos_bias, tab);

  // 2) QKV projection (12288 waves)
  qkv_gemm<<<(TOK / 16) * (D3 / 64) / 4, 128, 0, stream>>>(
      x16, qkvwT, qkv_b, k_delta, v_delta, q16, k16, vt16);

  // 3) gate projection (4096 waves)
  gate_gemm<<<(TOK / 16) * (DD / 64) / 4, 128, 0, stream>>>(
      x16, gatewT, gate_b, gate16);

  // 4) flash attention (4096 waves)
  attn_kernel<<<BB * HH * (NN / 16) / 4, 128, 0, stream>>>(
      q16, k16, vt16, tab, if_gain, y16);

  // 5) output projection -> d_out f32
  out_gemm<<<(TOK / 16) * (DD / 64) / 4, 128, 0, stream>>>(
      y16, gate16, outwT, out_b, (float*)d_out);
}
